// Decoder_85109071937993
// MI455X (gfx1250) — matrix operand order repaired and checked by reading, not compiled, not run on hardware
//
#include <hip/hip_runtime.h>
#include <hip/hip_bf16.h>
#include <math.h>

typedef __attribute__((ext_vector_type(16))) _Float16 v16h;
typedef __attribute__((ext_vector_type(8)))  _Float16 v8h;
typedef __attribute__((ext_vector_type(8)))  float    v8f;
typedef __attribute__((ext_vector_type(4)))  int      v4i;

#define LNUM  6
#define EDIM  384
#define SEQ   256
#define NH    6
#define HD    64
#define VOC   65
#define FFD   1536
#define BATCH 64
#define MTOK  (BATCH*SEQ)   // 16384 tokens

// ---------------------------------------------------------------------------
// gfx1250 async global->LDS copy (ASYNCcnt-tracked), with safe fallback
// ---------------------------------------------------------------------------
#if __has_builtin(__builtin_amdgcn_global_load_async_to_lds_b128)
#define HAVE_ASYNC_LD 1
typedef __attribute__((address_space(1))) v4i as1_v4i;
typedef __attribute__((address_space(3))) v4i as3_v4i;
__device__ __forceinline__ void async_cp16B(const _Float16* g, _Float16* l) {
  __builtin_amdgcn_global_load_async_to_lds_b128(
      (as1_v4i*)(unsigned long long)g,
      (as3_v4i*)(unsigned int)(unsigned long long)l, 0, 0);
}
__device__ __forceinline__ void wait_async0() {
#if __has_builtin(__builtin_amdgcn_s_wait_asynccnt)
  __builtin_amdgcn_s_wait_asynccnt(0);
#else
  asm volatile("s_wait_asynccnt 0" ::: "memory");
#endif
}
#endif

// ---------------------------------------------------------------------------
// WMMA fragment helpers (contiguous k per lane half, same order in A and B, wave32)
// ---------------------------------------------------------------------------
__device__ __forceinline__ v8f wmma32(v16h a, v16h b, v8f c) {
  return __builtin_amdgcn_wmma_f32_16x16x32_f16(false, a, false, b, (short)0, c,
                                                false, false);
}

// A (16x32, f16): lane L holds row (L&15); lanes<16: K=[k0,k0+16),
// lanes>=16: K=[k0+16,k0+32)
__device__ __forceinline__ v16h a_frag(const _Float16* __restrict__ base,
                                       int stride, int row, int kofs, int lane) {
  int r  = row + (lane & 15);
  int kb = kofs + ((lane & 16) ? 16 : 0);
  const _Float16* p = base + (long)r * stride + kb;
  v8h lo = *(const v8h*)p;
  v8h hi = *(const v8h*)(p + 8);
  v16h a;
#pragma unroll
  for (int i = 0; i < 8; i++) { a[i] = lo[i]; a[i + 8] = hi[i]; }
  return a;
}

// B (32x16, f16) from transposed storage Bt[n][k]: lane L holds column (L&15),
// contiguous 16 K values starting at k0 (+16 for lanes>=16)
__device__ __forceinline__ v16h b_frag(const _Float16* __restrict__ base,
                                       long stride, int col, long kofs, int lane) {
  int  c  = col + (lane & 15);
  long kb = kofs + ((lane & 16) ? 16 : 0);
  return *(const v16h*)(base + (long)c * stride + kb);
}

// ---------------------------------------------------------------------------
// Embedding: h[b,t,:] = tok_emb[x[b,t]] + pos_emb[t]
// ---------------------------------------------------------------------------
__global__ void embed_k(const int* __restrict__ x, const float* __restrict__ tok,
                        const float* __restrict__ pos, float* __restrict__ h) {
  long i = (long)blockIdx.x * blockDim.x + threadIdx.x;
  if (i >= (long)MTOK * EDIM) return;
  int  e  = (int)(i % EDIM);
  long t  = i / EDIM;
  int  tt = (int)(t % SEQ);
  h[i] = tok[(long)x[t] * EDIM + e] + pos[(long)tt * EDIM + e];
}

// ---------------------------------------------------------------------------
// Weight convert + transpose: src fp32 [K,N] -> dst f16 [N,K]
// ---------------------------------------------------------------------------
__global__ void wcvt_k(const float* __restrict__ src, _Float16* __restrict__ dst,
                       int K, int N) {
  long i = (long)blockIdx.x * blockDim.x + threadIdx.x;
  if (i >= (long)K * N) return;
  int n = (int)(i % N);
  int k = (int)(i / N);
  dst[(long)n * K + k] = (_Float16)src[i];
}

// ---------------------------------------------------------------------------
// LayerNorm: fp32 row -> f16 row.  invdiv = 1/(E-ddof)
// ---------------------------------------------------------------------------
__global__ void ln_k(const float* __restrict__ x, const float* __restrict__ g,
                     const float* __restrict__ b, _Float16* __restrict__ y,
                     float invdiv) {
  __shared__ float red[128];
  int row = blockIdx.x;
  const float* xr = x + (long)row * EDIM;
  float s = 0.f;
  for (int e = threadIdx.x; e < EDIM; e += 128) s += xr[e];
  red[threadIdx.x] = s; __syncthreads();
  for (int o = 64; o > 0; o >>= 1) {
    if ((int)threadIdx.x < o) red[threadIdx.x] += red[threadIdx.x + o];
    __syncthreads();
  }
  float mean = red[0] * (1.0f / EDIM);
  __syncthreads();
  float s2 = 0.f;
  for (int e = threadIdx.x; e < EDIM; e += 128) {
    float d = xr[e] - mean; s2 += d * d;
  }
  red[threadIdx.x] = s2; __syncthreads();
  for (int o = 64; o > 0; o >>= 1) {
    if ((int)threadIdx.x < o) red[threadIdx.x] += red[threadIdx.x + o];
    __syncthreads();
  }
  float rstd = rsqrtf(red[0] * invdiv + 1e-5f);
  for (int e = threadIdx.x; e < EDIM; e += 128)
    y[(long)row * EDIM + e] = (_Float16)(g[e] * (xr[e] - mean) * rstd + b[e]);
}

// ---------------------------------------------------------------------------
// WMMA GEMM: C[M,N] = A[M,K](f16) x Bt[N,K](f16) (+bias)
// MODE 0: f16 out | 1: f16 out + ReLU | 2: fp32 residual += | 3: f16 transposed
// K is compile-time (384 or 1536) so the k-loop unrolls/pipelines.
// Block = 256 threads (8 waves); tile = 256 rows x 64 cols; wave w owns rows
// [w*16, w*16+16) and [128+w*16, ...) -> 8 accumulators, 2x B-tile reuse.
// B tile (64x32 halves) staged in LDS, double-buffered, async-to-LDS on gfx1250.
// ---------------------------------------------------------------------------
#define BROW 48   // padded LDS row length (halves); 96B = 3x32B -> aligned v16h

template <int MODE, int K>
__global__ void gemm_k(const _Float16* __restrict__ A,
                       const _Float16* __restrict__ Bt,
                       const float* __restrict__ bias,
                       _Float16* __restrict__ O16, float* __restrict__ O32,
                       int M, int N) {
  __shared__ __align__(32) _Float16 Bl[2][64 * BROW];
  int tid  = threadIdx.x;
  int lane = tid & 31;
  int wave = tid >> 5;
  int m0 = blockIdx.y * 256 + wave * 16;   // second row group at m0 + 128
  int n0 = blockIdx.x * 64;

  // staging: thread t copies 16 bytes: col c = t/4, k-chunk (t%4)*8
  int sc = tid >> 2;
  int sk = (tid & 3) * 8;
  const _Float16* sbase = Bt + (long)(n0 + sc) * K + sk;

  v8f acc[2][4] = {};
  const int nk = K / 32;

  // prefetch tile 0
#ifdef HAVE_ASYNC_LD
  async_cp16B(sbase, &Bl[0][sc * BROW + sk]);
#else
  *(v8h*)&Bl[0][sc * BROW + sk] = *(const v8h*)sbase;
#endif

  for (int kk = 0; kk < nk; kk++) {
#ifdef HAVE_ASYNC_LD
    wait_async0();
#endif
    __syncthreads();
    if (kk + 1 < nk) {
      const _Float16* s = sbase + (kk + 1) * 32;
      _Float16* d = &Bl[(kk + 1) & 1][sc * BROW + sk];
#ifdef HAVE_ASYNC_LD
      async_cp16B(s, d);
#else
      *(v8h*)d = *(const v8h*)s;
#endif
    }
    int buf = kk & 1;
    v16h a0 = a_frag(A, K, m0,       kk * 32, lane);
    v16h a1 = a_frag(A, K, m0 + 128, kk * 32, lane);
#pragma unroll
    for (int f = 0; f < 4; f++) {
      int cc = f * 16 + (lane & 15);
      int kb = (lane & 16) ? 16 : 0;
      v16h b = *(const v16h*)&Bl[buf][cc * BROW + kb];
      acc[0][f] = wmma32(a0, b, acc[0][f]);
      acc[1][f] = wmma32(a1, b, acc[1][f]);
    }
  }

  int n = lane & 15;
  int rbase = (lane & 16) ? 8 : 0;
#pragma unroll
  for (int g = 0; g < 2; g++) {
#pragma unroll
    for (int f = 0; f < 4; f++) {
      int col = n0 + f * 16 + n;
      float bv = bias ? bias[col] : 0.f;
#pragma unroll
      for (int r = 0; r < 8; r++) {
        int row = m0 + g * 128 + rbase + r;
        float v = acc[g][f][r] + bv;
        if (MODE == 1) v = v > 0.f ? v : 0.f;
        if (MODE == 0 || MODE == 1)      O16[(long)row * N + col] = (_Float16)v;
        else if (MODE == 2)              O32[(long)row * N + col] += v;
        else                             O16[(long)col * M + row] = (_Float16)v;
      }
    }
  }
}

// ---------------------------------------------------------------------------
// Flash attention: 1 wave per (batch, head, 16-query tile).
// Wave-uniform running max (softmax is shift-invariant; any shift >= rowmax is
// numerically safe), row sums via P @ ones WMMA so they land in C-frag layout.
// q16/k16: [tok, E] f16.  vT: [E, MTOK] f16 (head-dim major).  att: [tok,E] f16
// ---------------------------------------------------------------------------
__global__ void attn_k(const _Float16* __restrict__ q16,
                       const _Float16* __restrict__ k16,
                       const _Float16* __restrict__ vT,
                       _Float16* __restrict__ att) {
  __shared__ __align__(64) _Float16 Pl[16 * 32];
  int lane = threadIdx.x;
  int blk = blockIdx.x;
  int qt = (blk % (SEQ / 16)) * 16;
  int hh = (blk / (SEQ / 16)) % NH;
  int bb = blk / ((SEQ / 16) * NH);
  const float scale = 0.125f;  // D^-0.5, D=64
  int rowg  = bb * SEQ + qt;
  int n     = lane & 15;
  int rbase = (lane & 16) ? 8 : 0;

  v16h qa0 = a_frag(q16, EDIM, rowg, hh * HD + 0,  lane);
  v16h qa1 = a_frag(q16, EDIM, rowg, hh * HD + 32, lane);

  v16h ones;
#pragma unroll
  for (int i = 0; i < 16; i++) ones[i] = (_Float16)1.0f;

  v8f o[4] = {};
  v8f lacc = {};            // per-row softmax denominators (all cols identical)
  float mrun = -3.0e38f;    // wave-uniform running max

  for (int kt = 0; kt <= qt; kt += 32) {
    // issue all four K-fragment loads, then the WMMA chain (one wait, not 4)
    v16h b0 = b_frag(k16, EDIM, bb * SEQ + kt,      hh * HD + 0,  lane);
    v16h b1 = b_frag(k16, EDIM, bb * SEQ + kt,      hh * HD + 32, lane);
    v16h b2 = b_frag(k16, EDIM, bb * SEQ + kt + 16, hh * HD + 0,  lane);
    v16h b3 = b_frag(k16, EDIM, bb * SEQ + kt + 16, hh * HD + 32, lane);
    v8f s0 = {}, s1 = {};
    s0 = wmma32(qa0, b0, s0);
    s0 = wmma32(qa1, b1, s0);
    s1 = wmma32(qa0, b2, s1);
    s1 = wmma32(qa1, b3, s1);

    float a0[8], a1[8];
    float lm = -3.0e38f;
#pragma unroll
    for (int r = 0; r < 8; r++) {
      int rg = qt + rbase + r;
      a0[r] = (kt + n      <= rg) ? s0[r] * scale : -3.0e38f;
      a1[r] = (kt + 16 + n <= rg) ? s1[r] * scale : -3.0e38f;
      lm = fmaxf(lm, fmaxf(a0[r], a1[r]));
    }
#pragma unroll
    for (int off = 16; off >= 1; off >>= 1)       // full-wave max, 5 steps
      lm = fmaxf(lm, __shfl_xor(lm, off, 32));
    float mn = fmaxf(mrun, lm);
    float sc = expf(mrun - mn);
    mrun = mn;
#pragma unroll
    for (int r = 0; r < 8; r++) {
      Pl[(rbase + r) * 32 + n]      = (_Float16)expf(a0[r] - mn);
      Pl[(rbase + r) * 32 + 16 + n] = (_Float16)expf(a1[r] - mn);
    }
#pragma unroll
    for (int f = 0; f < 4; f++)
#pragma unroll
      for (int r = 0; r < 8; r++) o[f][r] *= sc;
#pragma unroll
    for (int r = 0; r < 8; r++) lacc[r] *= sc;
    __syncthreads();
    // transpose C-layout P (in LDS) into an A-fragment
    v16h pa;
    {
      int lr = lane & 15;
      int kb = (lane & 16) ? 16 : 0;
      v8h lo = *(const v8h*)&Pl[lr * 32 + kb];
      v8h hi = *(const v8h*)&Pl[lr * 32 + kb + 8];
#pragma unroll
      for (int i = 0; i < 8; i++) { pa[i] = lo[i]; pa[i + 8] = hi[i]; }
    }
    __syncthreads();
    // all four V-fragment loads first, then the WMMA chain
    v16h bv0 = b_frag(vT, MTOK, hh * HD + 0,  bb * SEQ + kt, lane);
    v16h bv1 = b_frag(vT, MTOK, hh * HD + 16, bb * SEQ + kt, lane);
    v16h bv2 = b_frag(vT, MTOK, hh * HD + 32, bb * SEQ + kt, lane);
    v16h bv3 = b_frag(vT, MTOK, hh * HD + 48, bb * SEQ + kt, lane);
    lacc = wmma32(pa, ones, lacc);                // row sums, same frag layout
    o[0] = wmma32(pa, bv0, o[0]);
    o[1] = wmma32(pa, bv1, o[1]);
    o[2] = wmma32(pa, bv2, o[2]);
    o[3] = wmma32(pa, bv3, o[3]);
  }
#pragma unroll
  for (int f = 0; f < 4; f++)
#pragma unroll
    for (int r = 0; r < 8; r++) {
      int row = rowg + rbase + r;
      att[(long)row * EDIM + hh * HD + f * 16 + n] =
          (_Float16)(o[f][r] / lacc[r]);
    }
}

// ---------------------------------------------------------------------------
// LM head: out[t,v] = blm[v] + sum_e hn[t,e] * Wlm[e,v]   (V=65, tiny)
// ---------------------------------------------------------------------------
__global__ void lmhead_k(const _Float16* __restrict__ hn,
                         const float* __restrict__ Wlm,
                         const float* __restrict__ blm, float* __restrict__ out) {
  long i = (long)blockIdx.x * blockDim.x + threadIdx.x;
  if (i >= (long)MTOK * VOC) return;
  int  v = (int)(i % VOC);
  long t = i / VOC;
  const _Float16* hr = hn + t * EDIM;
  float acc = blm[v];
  for (int e = 0; e < EDIM; e++) acc += (float)hr[e] * Wlm[(long)e * VOC + v];
  out[i] = acc;
}

// ---------------------------------------------------------------------------
extern "C" void kernel_launch(void* const* d_in, const int* in_sizes, int n_in,
                              void* d_out, int out_size, void* d_ws, size_t ws_size,
                              hipStream_t stream) {
  (void)in_sizes; (void)n_in; (void)out_size; (void)ws_size;
  const int*   x     = (const int*)d_in[0];
  const float* tok   = (const float*)d_in[1];
  const float* pos   = (const float*)d_in[2];
  const float* Wq    = (const float*)d_in[3];
  const float* Wk    = (const float*)d_in[4];
  const float* Wv    = (const float*)d_in[5];
  const float* Wproj = (const float*)d_in[6];
  const float* bproj = (const float*)d_in[7];
  const float* W1    = (const float*)d_in[8];
  const float* b1    = (const float*)d_in[9];
  const float* W2    = (const float*)d_in[10];
  const float* b2    = (const float*)d_in[11];
  const float* ln1g  = (const float*)d_in[12];
  const float* ln1b  = (const float*)d_in[13];
  const float* ln2g  = (const float*)d_in[14];
  const float* ln2b  = (const float*)d_in[15];
  const float* lnfg  = (const float*)d_in[16];
  const float* lnfb  = (const float*)d_in[17];
  const float* Wlm   = (const float*)d_in[18];
  const float* blm   = (const float*)d_in[19];
  float* out = (float*)d_out;

  char* ws = (char*)d_ws;
  size_t off = 0;
  auto alloc = [&](size_t bytes) -> char* {
    char* p = ws + off;
    off += (bytes + 255) & ~(size_t)255;
    return p;
  };
  float*    h     = (float*)   alloc((size_t)MTOK * EDIM * 4);
  _Float16* hn16  = (_Float16*)alloc((size_t)MTOK * EDIM * 2);
  _Float16* q16   = (_Float16*)alloc((size_t)MTOK * EDIM * 2);
  _Float16* k16   = (_Float16*)alloc((size_t)MTOK * EDIM * 2);
  _Float16* vT16  = (_Float16*)alloc((size_t)MTOK * EDIM * 2);
  _Float16* att16 = (_Float16*)alloc((size_t)MTOK * EDIM * 2);
  _Float16* ff16  = (_Float16*)alloc((size_t)MTOK * FFD * 2);
  _Float16* WqT   = (_Float16*)alloc((size_t)LNUM * EDIM * EDIM * 2);
  _Float16* WkT   = (_Float16*)alloc((size_t)LNUM * EDIM * EDIM * 2);
  _Float16* WvT   = (_Float16*)alloc((size_t)LNUM * EDIM * EDIM * 2);
  _Float16* WpT   = (_Float16*)alloc((size_t)LNUM * EDIM * EDIM * 2);
  _Float16* W1T   = (_Float16*)alloc((size_t)LNUM * EDIM * FFD * 2);
  _Float16* W2T   = (_Float16*)alloc((size_t)LNUM * FFD * EDIM * 2);

  // -- weight conversion (transposed, f16) --
  const long ee = (long)EDIM * EDIM, ef = (long)EDIM * FFD;
  dim3 gEE((unsigned)((ee + 255) / 256)), gEF((unsigned)((ef + 255) / 256));
  for (int l = 0; l < LNUM; l++) {
    wcvt_k<<<gEE, 256, 0, stream>>>(Wq    + l * ee, WqT + l * ee, EDIM, EDIM);
    wcvt_k<<<gEE, 256, 0, stream>>>(Wk    + l * ee, WkT + l * ee, EDIM, EDIM);
    wcvt_k<<<gEE, 256, 0, stream>>>(Wv    + l * ee, WvT + l * ee, EDIM, EDIM);
    wcvt_k<<<gEE, 256, 0, stream>>>(Wproj + l * ee, WpT + l * ee, EDIM, EDIM);
    wcvt_k<<<gEF, 256, 0, stream>>>(W1    + l * ef, W1T + l * ef, EDIM, FFD);
    wcvt_k<<<gEF, 256, 0, stream>>>(W2    + l * ef, W2T + l * ef, FFD, EDIM);
  }

  // -- embedding --
  {
    long tot = (long)MTOK * EDIM;
    embed_k<<<(unsigned)((tot + 255) / 256), 256, 0, stream>>>(x, tok, pos, h);
  }

  const float inv_unbiased = 1.0f / (EDIM - 1);  // ddof=1
  const float inv_biased   = 1.0f / EDIM;        // ddof=0
  dim3 gE(EDIM / 64, MTOK / 256);                // N=384 GEMMs
  dim3 gF(FFD / 64, MTOK / 256);                 // N=1536 GEMM
  unsigned attnBlocks = BATCH * NH * (SEQ / 16); // 6144

  for (int l = 0; l < LNUM; l++) {
    // LN1 -> hn16
    ln_k<<<MTOK, 128, 0, stream>>>(h, ln1g + l * EDIM, ln1b + l * EDIM, hn16,
                                   inv_unbiased);
    // QKV
    gemm_k<0, EDIM><<<gE, 256, 0, stream>>>(hn16, WqT + l * ee, nullptr, q16,
                                            nullptr, MTOK, EDIM);
    gemm_k<0, EDIM><<<gE, 256, 0, stream>>>(hn16, WkT + l * ee, nullptr, k16,
                                            nullptr, MTOK, EDIM);
    gemm_k<3, EDIM><<<gE, 256, 0, stream>>>(hn16, WvT + l * ee, nullptr, vT16,
                                            nullptr, MTOK, EDIM);
    // attention
    attn_k<<<attnBlocks, 32, 0, stream>>>(q16, k16, vT16, att16);
    // proj + residual into h
    gemm_k<2, EDIM><<<gE, 256, 0, stream>>>(att16, WpT + l * ee, bproj + l * EDIM,
                                            nullptr, h, MTOK, EDIM);
    // LN2 -> hn16
    ln_k<<<MTOK, 128, 0, stream>>>(h, ln2g + l * EDIM, ln2b + l * EDIM, hn16,
                                   inv_unbiased);
    // FF1 (ReLU) -> ff16 ; FF2 + residual into h
    gemm_k<1, EDIM><<<gF, 256, 0, stream>>>(hn16, W1T + l * ef, b1 + l * FFD,
                                            ff16, nullptr, MTOK, FFD);
    gemm_k<2, FFD><<<gE, 256, 0, stream>>>(ff16, W2T + l * ef, b2 + l * EDIM,
                                           nullptr, h, MTOK, EDIM);
  }

  // final LN (biased var) + LM head
  ln_k<<<MTOK, 128, 0, stream>>>(h, lnfg, lnfb, hn16, inv_biased);
  {
    long tot = (long)MTOK * VOC;
    lmhead_k<<<(unsigned)((tot + 255) / 256), 256, 0, stream>>>(hn16, Wlm, blm, out);
  }
}
